// GNNLayer_16355235463442
// MI455X (gfx1250) — matrix operand order repaired and checked by reading, not compiled, not run on hardware
//
#include <hip/hip_runtime.h>
#include <hip/hip_bf16.h>

// ---------------------------------------------------------------------------
// GNN layer for MI455X (gfx1250, wave32):
//   Lf  = spmm(val,row,col, X)
//   Li  = spmm(val,row,col, Lf*X)
//   out = (Lf+X)@W1 + Li@W2 + (b1+b2)
//       = [Lf+X | Li] (N x 256, bf16) @ [W1;W2] (256 x 128, bf16)  + bias
// GEMM runs on v_wmma_f32_16x16x32_bf16 with fragments pre-packed in global
// memory as bf16 k-pairs (16 contiguous k per lane half), so fragment loads are global_load_b128.
// ---------------------------------------------------------------------------

typedef __attribute__((ext_vector_type(16))) __bf16 v16bf;
typedef __attribute__((ext_vector_type(8)))  float  v8f;

#define FDIM 128

__device__ __forceinline__ unsigned bf16_rne(float x) {
    unsigned u = __float_as_uint(x);
    return (u + 0x7FFFu + ((u >> 16) & 1u)) >> 16;
}
__device__ __forceinline__ unsigned pack2(float lo, float hi) {
    return bf16_rne(lo) | (bf16_rne(hi) << 16);
}

// ---------------- zero workspace (capture-safe) ----------------
__global__ void zero_f4(float* __restrict__ p, long n4) {
    long i = (long)blockIdx.x * blockDim.x + threadIdx.x;
    long stride = (long)gridDim.x * blockDim.x;
    float4 z = make_float4(0.f, 0.f, 0.f, 0.f);
    for (; i < n4; i += stride) ((float4*)p)[i] = z;
}

// ---------------- SpMM pass 1: Lf[row] += val * X[col] ----------------
// one wave32 per edge; lane owns 4 consecutive features (float4).
__global__ void spmm_feat(const float* __restrict__ feat,
                          const float* __restrict__ val,
                          const int*  __restrict__ rows,
                          const int*  __restrict__ cols,
                          float* __restrict__ out, int nnz) {
    const int lane = threadIdx.x & 31;
    int wid = blockIdx.x * (blockDim.x >> 5) + (threadIdx.x >> 5);
    const int nw = gridDim.x * (blockDim.x >> 5);
    for (int e = wid; e < nnz; e += nw) {
        const float v = val[e];           // uniform per wave -> scalarized
        const long  c = cols[e];
        const long  r = rows[e];
        const float4 f = *(const float4*)(feat + c * FDIM + lane * 4);
        float* dst = out + r * FDIM + lane * 4;
        atomicAdd(dst + 0, v * f.x);
        atomicAdd(dst + 1, v * f.y);
        atomicAdd(dst + 2, v * f.z);
        atomicAdd(dst + 3, v * f.w);
    }
}

// ------- SpMM pass 2: Li[row] += val * (Lf[col]*X[col])  (fused gather) -----
__global__ void spmm_inter(const float* __restrict__ feat,
                           const float* __restrict__ Lf,
                           const float* __restrict__ val,
                           const int*  __restrict__ rows,
                           const int*  __restrict__ cols,
                           float* __restrict__ out, int nnz) {
    const int lane = threadIdx.x & 31;
    int wid = blockIdx.x * (blockDim.x >> 5) + (threadIdx.x >> 5);
    const int nw = gridDim.x * (blockDim.x >> 5);
    for (int e = wid; e < nnz; e += nw) {
        const float v = val[e];
        const long  c = cols[e];
        const long  r = rows[e];
        const float4 f = *(const float4*)(feat + c * FDIM + lane * 4);
        const float4 g = *(const float4*)(Lf   + c * FDIM + lane * 4);
        float* dst = out + r * FDIM + lane * 4;
        atomicAdd(dst + 0, v * f.x * g.x);
        atomicAdd(dst + 1, v * f.y * g.y);
        atomicAdd(dst + 2, v * f.z * g.z);
        atomicAdd(dst + 3, v * f.w * g.w);
    }
}

// ------- pack [W1;W2] (256x128 f32, k-major) -> Wpk[n][kpair] bf16x2 -------
// B-fragment friendly: lane's 8 k-pairs at fixed n are 8 contiguous u32.
__global__ void pack_w(const float* __restrict__ W1,
                       const float* __restrict__ W2,
                       unsigned* __restrict__ Wpk) {
    const int idx = blockIdx.x * blockDim.x + threadIdx.x;   // 0..16383
    const int n  = idx >> 7;
    const int kp = idx & 127;
    float a, b;
    if (kp < 64) { const int k = kp * 2;       a = W1[k * FDIM + n]; b = W1[(k + 1) * FDIM + n]; }
    else         { const int k = kp * 2 - 128; a = W2[k * FDIM + n]; b = W2[(k + 1) * FDIM + n]; }
    Wpk[idx] = pack2(a, b);
}

// ------- pack A = [Lf+X | Li] (N x 256 f32) -> Apk[row][kpair] bf16x2 -------
__global__ void pack_a(const float* __restrict__ feat,
                       const float* __restrict__ Lf,
                       const float* __restrict__ Li,
                       unsigned* __restrict__ Apk, int n_rows) {
    const long total = (long)n_rows * FDIM;
    long i = (long)blockIdx.x * blockDim.x + threadIdx.x;
    const long stride = (long)gridDim.x * blockDim.x;
    for (; i < total; i += stride) {
        const long row = i >> 7;
        const int  kp  = (int)(i & 127);
        float a, b;
        if (kp < 64) {
            const long o = row * FDIM + kp * 2;
            a = Lf[o]     + feat[o];
            b = Lf[o + 1] + feat[o + 1];
        } else {
            const long o = row * FDIM + (kp * 2 - 128);
            a = Li[o];
            b = Li[o + 1];
        }
        Apk[i] = pack2(a, b);
    }
}

// ---------------- fused GEMM: out = Apk @ Wpk + (b1+b2) ----------------
// 256 threads = 8 waves. Wave w -> 16-row output strip, full 128 cols
// (8 tiles of 16x16). K = 256 (8 chunks of 32).
// A fragment (16x32): lanes 0-15 hold K 0-15 of row M=lane,
// lanes 16-31 hold K 16-31 -> two global_load_b128 per lane per chunk.
// B layout (32x16): lanes 0-15 K=0-15, lanes 16-31 K=16-31 at N=lane&15
// -> two contiguous global_load_b128 from Wpk[n][kpair].
// Out-of-range A rows are CLAMPED (not zeroed): an A row only contaminates
// its own output row, and out-of-range output rows are never stored. This
// keeps every fragment load an unconditional global_load_b128 (no cndmask /
// flat path) and EXEC all-ones through every WMMA.
__global__ void gemm_wmma(const unsigned* __restrict__ Apk,
                          const unsigned* __restrict__ Wpk,
                          const float* __restrict__ b1,
                          const float* __restrict__ b2,
                          float* __restrict__ out, int n_rows) {
    const int lane = threadIdx.x & 31;
    const int wave = threadIdx.x >> 5;
    const int half = lane >> 4;      // K-half selector for A/B fragments
    const int nlo  = lane & 15;

    const long row0 = (long)blockIdx.x * 128 + wave * 16;
    long arow = row0 + nlo;                          // A frag row for this lane
    if (arow >= n_rows) arow = n_rows - 1;           // clamp: see note above
    const uint4* Ab = (const uint4*)(Apk + arow * FDIM);   // 128 u32 per row

    float bias[8];
#pragma unroll
    for (int nt = 0; nt < 8; ++nt) {
        const int n = nt * 16 + nlo;
        bias[nt] = b1[n] + b2[n];
    }

    const v8f zacc = {0.f, 0.f, 0.f, 0.f, 0.f, 0.f, 0.f, 0.f};
    v8f acc[8];
#pragma unroll
    for (int nt = 0; nt < 8; ++nt) acc[nt] = zacc;

#pragma unroll
    for (int kc = 0; kc < 8; ++kc) {
        // A fragment: kpairs kc*16 + half*8 + [0..7]
        union { uint4 u[2]; v16bf v; } af;
        af.u[0] = Ab[kc * 4 + 2 * half];              // uint4 index = u32/4
        af.u[1] = Ab[kc * 4 + 2 * half + 1];

#pragma unroll
        for (int nt = 0; nt < 8; ++nt) {
            // B fragment: 8 contiguous kpairs at n = nt*16 + nlo
            const uint4* Bb =
                (const uint4*)(Wpk + (nt * 16 + nlo) * FDIM + kc * 16 + half * 8);
            union { uint4 u[2]; v16bf v; } bf;
            bf.u[0] = Bb[0];
            bf.u[1] = Bb[1];
            acc[nt] = __builtin_amdgcn_wmma_f32_16x16x32_bf16(
                false, af.v, false, bf.v, (short)0, acc[nt], false, false);
        }
    }

    // C/D layout: lanes 0-15: N=lane, VGPR v -> M=v ; lanes 16-31 -> M=8+v.
#pragma unroll
    for (int nt = 0; nt < 8; ++nt) {
#pragma unroll
        for (int v = 0; v < 8; ++v) {
            const long r = row0 + half * 8 + v;
            if (r < n_rows)
                out[r * FDIM + nt * 16 + nlo] = acc[nt][v] + bias[nt];
        }
    }
}

// ---------------------------------------------------------------------------
extern "C" void kernel_launch(void* const* d_in, const int* in_sizes, int n_in,
                              void* d_out, int out_size, void* d_ws, size_t ws_size,
                              hipStream_t stream) {
    const int N   = in_sizes[0] / FDIM;
    const int NNZ = in_sizes[1];

    const float* feat = (const float*)d_in[0];
    const float* eval = (const float*)d_in[1];
    const float* W1   = (const float*)d_in[2];
    const float* b1   = (const float*)d_in[3];
    const float* W2   = (const float*)d_in[4];
    const float* b2   = (const float*)d_in[5];
    const int*   erow = (const int*)d_in[6];
    const int*   ecol = (const int*)d_in[7];
    float* out = (float*)d_out;

    // workspace: Lf (N*128 f32) | Li (N*128 f32) | Apk (N*128 u32) | Wpk (16K u32)
    float*    Lf  = (float*)d_ws;
    float*    Li  = Lf + (size_t)N * FDIM;
    unsigned* Apk = (unsigned*)(Li + (size_t)N * FDIM);
    unsigned* Wpk = Apk + (size_t)N * FDIM;

    const long zero_n4 = (long)N * FDIM * 2 / 4;   // Lf and Li are contiguous
    zero_f4<<<2048, 256, 0, stream>>>(Lf, zero_n4);

    spmm_feat<<<2048, 256, 0, stream>>>(feat, eval, erow, ecol, Lf, NNZ);
    spmm_inter<<<2048, 256, 0, stream>>>(feat, Lf, eval, erow, ecol, Li, NNZ);

    pack_w<<<64, 256, 0, stream>>>(W1, W2, Wpk);
    pack_a<<<2048, 256, 0, stream>>>(feat, Lf, Li, Apk, N);

    gemm_wmma<<<(N + 127) / 128, 256, 0, stream>>>(Apk, Wpk, b1, b2, out, N);
}
